// ST_Transformer_Block_35338990912063
// MI455X (gfx1250) — compile-verified, matrix operand order checked by reading, not run on hardware
//
#include <hip/hip_runtime.h>
#include <hip/hip_bf16.h>
#include <math.h>

// ---------------------------------------------------------------------------
// ST-Transformer block for MI455X (gfx1250, wave32, WMMA).
// Dense GEMMs: v_wmma_f32_16x16x32_f16, A/W tiles staged in padded LDS,
// inner loop = 4x ds_load_b128 + 1 wmma per 32-K step (no branches, no cvt).
// Weights pre-converted to f16 once per launch (GCN weight pre-transposed).
// Attention cores (19x19 spatial, 120x120 temporal) are VALU + LDS with
// online softmax. Memory-bound design: big intermediates touched once.
// ---------------------------------------------------------------------------

typedef _Float16 f16;
typedef __attribute__((ext_vector_type(16))) _Float16 v16h;
typedef __attribute__((ext_vector_type(8)))  _Float16 v8h;
typedef __attribute__((ext_vector_type(8)))  float    v8f;

#define NT_TOT 7680      // N*T = 64*120
#define J_     25
#define C_     128
#define DFF_   256
#define K_KEEP 17
#define K_DROP 8
#define LSLOW  19        // 1 + 17 + 1
#define TT     120
#define NJ_    1600      // N*J
#define HD     32
#define NHEADS 4
#define KPAD   8         // LDS row pad (halfs): dword stride = K/2+4 == 4 mod 64

// ============================= generic WMMA GEMM ===========================
// Y[M,N] = act( A[M,K] @ W^T + bias ), W pre-converted f16, (N,K) row-major.
// grid = (M/16, N/64), block = 128 (4 waves, one 16x16 tile each).
template<int ACT>
__global__ __launch_bounds__(128)
void gemm_wmma(const float* __restrict__ A, const f16* __restrict__ Wh,
               const float* __restrict__ bias, float* __restrict__ Y,
               int N, int K)
{
    __shared__ f16 Atile[16 * (384 + KPAD)];
    __shared__ f16 Wtile[64 * (384 + KPAD)];
    const int  KP   = K + KPAD;
    const int  tid  = threadIdx.x;
    const long m0   = (long)blockIdx.x * 16;
    const int  nblk = blockIdx.y * 64;

    // A: 16 x K (f32 -> f16), float4 global loads
    for (int idx = tid * 4; idx < 16 * K; idx += 512) {
        int r = idx / K, c = idx - r * K;
        const float4 f = *(const float4*)&A[(m0 + r) * (long)K + c];
        f16* dst = &Atile[r * KP + c];
        dst[0] = (f16)f.x; dst[1] = (f16)f.y; dst[2] = (f16)f.z; dst[3] = (f16)f.w;
    }
    // W: 64 x K (already f16), b128 global loads
    for (int idx = tid * 8; idx < 64 * K; idx += 1024) {
        int r = idx / K, c = idx - r * K;
        *(v8h*)&Wtile[r * KP + c] = *(const v8h*)&Wh[(long)(nblk + r) * K + c];
    }
    __syncthreads();

    const int wave = tid >> 5;
    const int lane = tid & 31;
    const int m    = lane & 15;                 // A row for this lane
    const int bn   = wave * 16 + (lane & 15);   // W row (output col) in block
    const int koff = (lane & 16) ? 8 : 0;       // ISA 16-bit fragment layout

    const f16* arow = &Atile[m * KP];
    const f16* brow = &Wtile[bn * KP];

    v8f acc = {0.f, 0.f, 0.f, 0.f, 0.f, 0.f, 0.f, 0.f};
    for (int kk = 0; kk < K; kk += 32) {
        // two contiguous, 16B-aligned 8-half runs per matrix == ds_load_b128 x4
        v8h alo = *(const v8h*)&arow[kk + koff];
        v8h ahi = *(const v8h*)&arow[kk + 16 + koff];
        v8h blo = *(const v8h*)&brow[kk + koff];
        v8h bhi = *(const v8h*)&brow[kk + 16 + koff];
        v16h a = __builtin_shufflevector(alo, ahi, 0,1,2,3,4,5,6,7,8,9,10,11,12,13,14,15);
        v16h b = __builtin_shufflevector(blo, bhi, 0,1,2,3,4,5,6,7,8,9,10,11,12,13,14,15);
        acc = __builtin_amdgcn_wmma_f32_16x16x32_f16(false, a, false, b,
                                                     (short)0, acc, false, false);
    }

    const int   nn    = nblk + bn;
    const float bv    = bias ? bias[nn] : 0.f;
    const int   rbase = (lane & 16) ? 8 : 0;
#pragma unroll
    for (int r = 0; r < 8; ++r) {
        float v = acc[r] + bv;
        if (ACT == 1) v = 0.5f * v * (1.f + erff(v * 0.70710678118f)); // exact gelu
        Y[(m0 + r + rbase) * (long)N + nn] = v;
    }
}

// ======================== weight pre-conversion ============================
__global__ void cvt_w(const float* __restrict__ W, f16* __restrict__ o, int n)
{
    int i = blockIdx.x * 256 + threadIdx.x;
    if (i < n) o[i] = (f16)W[i];
}
// GCN weight: stored (K=128, N=128); emit (N,K) so GEMM sees one layout.
__global__ void cvt_w_t128(const float* __restrict__ W, f16* __restrict__ o)
{
    int i = blockIdx.x * 256 + threadIdx.x;
    if (i < 128 * 128) { int nn = i >> 7, k = i & 127; o[i] = (f16)W[k * 128 + nn]; }
}

// ============================= row-wise norms ==============================
__global__ __launch_bounds__(128)
void rms_rows(const float* __restrict__ X, const float* __restrict__ w,
              float* __restrict__ Y)
{
    __shared__ float red[128];
    const long row = blockIdx.x;
    const int  c   = threadIdx.x;
    float x = X[row * C_ + c];
    red[c] = x * x;
    __syncthreads();
    for (int s = 64; s > 0; s >>= 1) { if (c < s) red[c] += red[c + s]; __syncthreads(); }
    float r = rsqrtf(red[0] * (1.f / C_) + 1e-6f);
    Y[row * C_ + c] = x * r * w[c];
}

__global__ __launch_bounds__(128)
void add_layernorm(float* __restrict__ X, const float* __restrict__ R,
                   const float* __restrict__ w, const float* __restrict__ b)
{
    __shared__ float red[128];
    const long row = blockIdx.x;
    const int  c   = threadIdx.x;
    float x = X[row * C_ + c] + R[row * C_ + c];
    red[c] = x;
    __syncthreads();
    for (int s = 64; s > 0; s >>= 1) { if (c < s) red[c] += red[c + s]; __syncthreads(); }
    float mean = red[0] * (1.f / C_);
    __syncthreads();
    float d = x - mean;
    red[c] = d * d;
    __syncthreads();
    for (int s = 64; s > 0; s >>= 1) { if (c < s) red[c] += red[c + s]; __syncthreads(); }
    float inv = rsqrtf(red[0] * (1.f / C_) + 1e-5f);
    X[row * C_ + c] = d * inv * w[c] + b[c];
}

__global__ void add_rows(float* __restrict__ Y, const float* __restrict__ X, long n)
{
    long i = (long)blockIdx.x * blockDim.x + threadIdx.x;
    if (i < n) Y[i] += X[i];
}

// ===================== pruning: stable descending argsort ==================
__global__ void sort_select(const float* __restrict__ scores,
                            int* __restrict__ idxk, int* __restrict__ idxd,
                            float* __restrict__ wd)
{
    int nt = blockIdx.x * blockDim.x + threadIdx.x;
    if (nt >= NT_TOT) return;
    float s[J_]; bool used[J_]; int ord[J_];
    for (int j = 0; j < J_; ++j) { s[j] = scores[(long)nt * J_ + j]; used[j] = false; }
    for (int p = 0; p < J_; ++p) {            // strict '>' => ties keep lowest index
        float best = -3.4e38f; int bi = 0;
        for (int j = 0; j < J_; ++j) if (!used[j] && s[j] > best) { best = s[j]; bi = j; }
        used[bi] = true; ord[p] = bi;
    }
    for (int i = 0; i < K_KEEP; ++i) idxk[nt * K_KEEP + i] = ord[i];
    float mx = -3.4e38f;
    for (int i = 0; i < K_DROP; ++i) {
        int j = ord[K_KEEP + i];
        idxd[nt * K_DROP + i] = j;
        mx = fmaxf(mx, s[j]);
    }
    float e[K_DROP], sum = 0.f;
    for (int i = 0; i < K_DROP; ++i) { e[i] = __expf(s[ord[K_KEEP + i]] - mx); sum += e[i]; }
    for (int i = 0; i < K_DROP; ++i) wd[nt * K_DROP + i] = e[i] / sum;
}

__global__ __launch_bounds__(128)
void build_xslow(const float* __restrict__ A, const int* __restrict__ idxk,
                 const int* __restrict__ idxd, const float* __restrict__ wd,
                 const float* __restrict__ cls, float* __restrict__ B)
{
    const int nt = blockIdx.x, c = threadIdx.x;
    const long ob = (long)nt * LSLOW * C_;
    B[ob + c] = cls[c];
    for (int i = 0; i < K_KEEP; ++i) {
        int j = idxk[nt * K_KEEP + i];
        B[ob + (1 + i) * C_ + c] = A[((long)nt * J_ + j) * C_ + c];
    }
    float rep = 0.f;
    for (int i = 0; i < K_DROP; ++i) {
        int j = idxd[nt * K_DROP + i];
        rep += wd[nt * K_DROP + i] * A[((long)nt * J_ + j) * C_ + c];
    }
    B[ob + (LSLOW - 1) * C_ + c] = rep;
}

// ========================== spatial attention core =========================
__global__ __launch_bounds__(32)
void attn_spatial(const float* __restrict__ QKV, float* __restrict__ O,
                  float* __restrict__ c2o)
{
    __shared__ float sq[LSLOW * 33], sk[LSLOW * 33], sv[LSLOW * 33], sp[32];
    const int  nt   = blockIdx.x;
    const int  lane = threadIdx.x;
    const long base = (long)nt * LSLOW;
    float c2oacc = 0.f;

    for (int h = 0; h < NHEADS; ++h) {
        __syncthreads();
        for (int i = lane; i < LSLOW * HD; i += 32) {
            int r = i >> 5, d = i & 31;
            long q = (base + r) * 384 + h * HD + d;
            sq[r * 33 + d] = QKV[q];
            sk[r * 33 + d] = QKV[q + 128];
            sv[r * 33 + d] = QKV[q + 256];
        }
        __syncthreads();
        for (int qi = 0; qi < LSLOW; ++qi) {
            float s = -3.4e38f;
            if (lane < LSLOW) {
                s = 0.f;
                for (int d = 0; d < HD; ++d) s += sq[qi * 33 + d] * sk[lane * 33 + d];
                s *= 0.176776695f;              // 1/sqrt(32)
            }
            sp[lane] = s;
            __syncthreads();
            float mx = sp[0];
            for (int j = 1; j < LSLOW; ++j) mx = fmaxf(mx, sp[j]);
            float e = (lane < LSLOW) ? __expf(s - mx) : 0.f;
            __syncthreads();
            sp[lane] = e;
            __syncthreads();
            float sum = 0.f;
            for (int j = 0; j < LSLOW; ++j) sum += sp[j];
            float o = 0.f;
            for (int j = 0; j < LSLOW; ++j) o += sp[j] * sv[j * 33 + lane];
            O[(base + qi) * C_ + h * HD + lane] = o / sum;
            if (qi == 0 && lane >= 1 && lane < LSLOW) c2oacc += e / sum;
            __syncthreads();
        }
    }
    if (lane >= 1 && lane < LSLOW)
        c2o[(long)nt * (LSLOW - 1) + lane - 1] = c2oacc * 0.25f;
}

// ========================= temporal attention core =========================
__global__ __launch_bounds__(128)
void attn_temporal(const float* __restrict__ QKV, float* __restrict__ O)
{
    __shared__ float sk[TT * 33], sv[TT * 33];
    const int  nj = blockIdx.x, h = blockIdx.y, tid = threadIdx.x;
    const long base = (long)nj * TT;
    for (int i = tid; i < TT * HD; i += 128) {
        int r = i >> 5, d = i & 31;
        long q = (base + r) * 384 + h * HD + d;
        sk[r * 33 + d] = QKV[q + 128];
        sv[r * 33 + d] = QKV[q + 256];
    }
    __syncthreads();
    if (tid < TT) {
        float qv[HD], o[HD];
#pragma unroll
        for (int d = 0; d < HD; ++d) { qv[d] = QKV[(base + tid) * 384 + h * HD + d]; o[d] = 0.f; }
        float m = -3.4e38f, l = 0.f;
        for (int j = 0; j < TT; ++j) {
            float s = 0.f;
#pragma unroll
            for (int d = 0; d < HD; ++d) s += qv[d] * sk[j * 33 + d];
            s *= 0.176776695f;
            float mn = fmaxf(m, s);
            float corr = __expf(m - mn);
            float e = __expf(s - mn);
            l = l * corr + e;
#pragma unroll
            for (int d = 0; d < HD; ++d) o[d] = o[d] * corr + e * sv[j * 33 + d];
            m = mn;
        }
#pragma unroll
        for (int d = 0; d < HD; ++d) O[(base + tid) * C_ + h * HD + d] = o[d] / l;
    }
}

// ======================= scatter back + build xt layout ====================
__global__ __launch_bounds__(128)
void scatter_xsp(const float* __restrict__ A, const float* __restrict__ B,
                 const float* __restrict__ Cx, const int* __restrict__ idxk,
                 const int* __restrict__ idxd, float* __restrict__ F,
                 float* __restrict__ G)
{
    const int nt = blockIdx.x, c = threadIdx.x;
    const int n = nt / TT, t = nt - n * TT;
    const long sb = (long)nt * LSLOW * C_;
    float rep = Cx[sb + (LSLOW - 1) * C_ + c] - B[sb + (LSLOW - 1) * C_ + c];
    for (int i = 0; i < K_KEEP; ++i) {
        int j = idxk[nt * K_KEEP + i];
        float v = Cx[sb + (1 + i) * C_ + c] + A[((long)nt * J_ + j) * C_ + c];
        F[((long)nt * J_ + j) * C_ + c] = v;
        G[(((long)n * J_ + j) * TT + t) * C_ + c] = v;
    }
    for (int i = 0; i < K_DROP; ++i) {
        int j = idxd[nt * K_DROP + i];
        float v = 2.f * A[((long)nt * J_ + j) * C_ + c] + rep;
        F[((long)nt * J_ + j) * C_ + c] = v;
        G[(((long)n * J_ + j) * TT + t) * C_ + c] = v;
    }
}

__global__ void write_next_attn(const int* __restrict__ idxk, const int* __restrict__ idxd,
                                const float* __restrict__ c2o, float* __restrict__ na)
{
    const int nt = blockIdx.x, t = threadIdx.x;  // blockDim = 32
    if (t < K_KEEP)
        na[(long)nt * J_ + idxk[nt * K_KEEP + t]] = c2o[(long)nt * 18 + t];
    else if (t < J_) {
        int i = t - K_KEEP;
        na[(long)nt * J_ + idxd[nt * K_DROP + i]] = c2o[(long)nt * 18 + K_KEEP];
    }
}

__global__ void final_add(const float* __restrict__ G, const float* __restrict__ F,
                          const float* __restrict__ x, float* __restrict__ out)
{
    long i = (long)blockIdx.x * blockDim.x + threadIdx.x;
    if (i >= (long)NT_TOT * J_ * C_) return;
    int  c   = i & (C_ - 1);
    long r   = i >> 7;            // (n*T + t)*J + j
    int  j   = (int)(r % J_);
    long ntq = r / J_;
    int  t   = (int)(ntq % TT);
    int  n   = (int)(ntq / TT);
    long g   = (((long)n * J_ + j) * TT + t) * C_ + c;
    out[i] = G[g] + F[i] + x[i];
}

// ================================ launcher =================================
extern "C" void kernel_launch(void* const* d_in, const int* in_sizes, int n_in,
                              void* d_out, int out_size, void* d_ws, size_t ws_size,
                              hipStream_t stream)
{
    (void)in_sizes; (void)n_in; (void)out_size; (void)ws_size;
    const float* x     = (const float*)d_in[0];
    const float* prev  = (const float*)d_in[1];
    const float* gcn_w = (const float*)d_in[2];   // gcn_w[0] only: diag(close/far)==0
    const float* gnw   = (const float*)d_in[3];
    const float* cls   = (const float*)d_in[4];
    const float* sn1w  = (const float*)d_in[5];
    const float* sn2w  = (const float*)d_in[6];
    const float* sinw  = (const float*)d_in[7];
    const float* sinb  = (const float*)d_in[8];
    const float* soutw = (const float*)d_in[9];
    const float* soutb = (const float*)d_in[10];
    const float* sf1w  = (const float*)d_in[11];
    const float* sf1b  = (const float*)d_in[12];
    const float* sf2w  = (const float*)d_in[13];
    const float* sf2b  = (const float*)d_in[14];
    const float* tinw  = (const float*)d_in[15];
    const float* tinb  = (const float*)d_in[16];
    const float* toutw = (const float*)d_in[17];
    const float* toutb = (const float*)d_in[18];
    const float* ln1w  = (const float*)d_in[19];
    const float* ln1b  = (const float*)d_in[20];
    const float* ln2w  = (const float*)d_in[21];
    const float* ln2b  = (const float*)d_in[22];
    const float* tf1w  = (const float*)d_in[23];
    const float* tf1b  = (const float*)d_in[24];
    const float* tf2w  = (const float*)d_in[25];
    const float* tf2b  = (const float*)d_in[26];
    const float* tnw   = (const float*)d_in[27];

    float* out_x  = (float*)d_out;                          // 24,576,000
    float* out_na = out_x + (long)NT_TOT * J_ * C_;         // 192,000

    // ---- workspace layout (floats); ~234.4M floats + 557KB f16 weights ----
    float* ws = (float*)d_ws;
    const long NJC  = (long)NT_TOT * J_ * C_;     // 24,576,000
    const long SLC  = (long)NT_TOT * LSLOW * C_;  // 18,677,760
    float* A_   = ws;                 // res_spatial (rmsnorm(gcn))
    float* B_   = A_  + NJC;          // x_slow
    float* C__  = B_  + SLC;          // x_msa -> x_final_slow
    float* Nrm  = C__ + SLC;          // normalized / temp rows (max 192000x128)
    float* QKV  = Nrm + NJC;          // qkv (max 192000x384); also FFN hidden
    float* Ho   = QKV + (long)192000 * 384;  // attention head output
    float* F_   = Ho  + NJC;          // x_sp  (N,T,J,C)
    float* G_   = F_  + NJC;          // xt    (N,J,T,C), processed in place
    int*   idxk = (int*)(G_ + NJC);
    int*   idxd = idxk + NT_TOT * K_KEEP;
    float* wd   = (float*)(idxd + NT_TOT * K_DROP);
    float* c2o  = wd + NT_TOT * K_DROP;

    // f16 weight copies (all offsets 16B aligned)
    f16* whG  = (f16*)(c2o + NT_TOT * 18);   // gcn (transposed)     16384
    f16* whSI = whG  + 16384;                // s_in_w               49152
    f16* whSO = whSI + 49152;                // s_out_w              16384
    f16* whF1 = whSO + 16384;                // s_f1w                32768
    f16* whF2 = whF1 + 32768;                // s_f2w                32768
    f16* whTI = whF2 + 32768;                // t_in_w               49152
    f16* whTO = whTI + 49152;                // t_out_w              16384
    f16* whG1 = whTO + 16384;                // t_f1w                32768
    f16* whG2 = whG1 + 32768;                // t_f2w                32768

    const long MROWS_ALL  = (long)NT_TOT * J_;      // 192000 rows
    const long MROWS_SLOW = (long)NT_TOT * LSLOW;   // 145920 rows

    // ---- weight pre-conversion (once per call) ----
    cvt_w_t128<<<64, 256, 0, stream>>>(gcn_w, whG);
    cvt_w<<<192, 256, 0, stream>>>(sinw,  whSI, 49152);
    cvt_w<<< 64, 256, 0, stream>>>(soutw, whSO, 16384);
    cvt_w<<<128, 256, 0, stream>>>(sf1w,  whF1, 32768);
    cvt_w<<<128, 256, 0, stream>>>(sf2w,  whF2, 32768);
    cvt_w<<<192, 256, 0, stream>>>(tinw,  whTI, 49152);
    cvt_w<<< 64, 256, 0, stream>>>(toutw, whTO, 16384);
    cvt_w<<<128, 256, 0, stream>>>(tf1w,  whG1, 32768);
    cvt_w<<<128, 256, 0, stream>>>(tf2w,  whG2, 32768);

    // ---------------- spatial branch ----------------
    gemm_wmma<0><<<dim3(MROWS_ALL / 16, 2), 128, 0, stream>>>(x, whG, nullptr, Nrm, 128, 128);
    rms_rows<<<(int)MROWS_ALL, 128, 0, stream>>>(Nrm, gnw, A_);

    sort_select<<<NT_TOT / 128, 128, 0, stream>>>(prev, idxk, idxd, wd);
    build_xslow<<<NT_TOT, 128, 0, stream>>>(A_, idxk, idxd, wd, cls, B_);

    rms_rows<<<(int)MROWS_SLOW, 128, 0, stream>>>(B_, sn1w, Nrm);
    gemm_wmma<0><<<dim3(MROWS_SLOW / 16, 6), 128, 0, stream>>>(Nrm, whSI, sinb, QKV, 384, 128);
    attn_spatial<<<NT_TOT, 32, 0, stream>>>(QKV, Ho, c2o);
    gemm_wmma<0><<<dim3(MROWS_SLOW / 16, 2), 128, 0, stream>>>(Ho, whSO, soutb, C__, 128, 128);
    add_rows<<<(int)(SLC / 256), 256, 0, stream>>>(C__, B_, SLC);               // x_msa

    rms_rows<<<(int)MROWS_SLOW, 128, 0, stream>>>(C__, sn2w, Nrm);
    gemm_wmma<1><<<dim3(MROWS_SLOW / 16, 4), 128, 0, stream>>>(Nrm, whF1, sf1b, QKV, 256, 128);
    gemm_wmma<0><<<dim3(MROWS_SLOW / 16, 2), 128, 0, stream>>>(QKV, whF2, sf2b, Ho, 128, 256);
    add_rows<<<(int)(SLC / 256), 256, 0, stream>>>(C__, Ho, SLC);               // x_final_slow

    scatter_xsp<<<NT_TOT, 128, 0, stream>>>(A_, B_, C__, idxk, idxd, F_, G_);
    write_next_attn<<<NT_TOT, 32, 0, stream>>>(idxk, idxd, c2o, out_na);

    // ---------------- temporal branch ----------------
    gemm_wmma<0><<<dim3(MROWS_ALL / 16, 6), 128, 0, stream>>>(G_, whTI, tinb, QKV, 384, 128);
    attn_temporal<<<dim3(NJ_, NHEADS), 128, 0, stream>>>(QKV, Ho);
    gemm_wmma<0><<<dim3(MROWS_ALL / 16, 2), 128, 0, stream>>>(Ho, whTO, toutb, Nrm, 128, 128);
    add_layernorm<<<(int)MROWS_ALL, 128, 0, stream>>>(G_, Nrm, ln1w, ln1b);     // xt = ln(xt + a)

    gemm_wmma<1><<<dim3(MROWS_ALL / 16, 4), 128, 0, stream>>>(G_, whG1, tf1b, QKV, 256, 128);
    gemm_wmma<0><<<dim3(MROWS_ALL / 16, 2), 128, 0, stream>>>(QKV, whG2, tf2b, Nrm, 128, 256);
    add_layernorm<<<(int)MROWS_ALL, 128, 0, stream>>>(G_, Nrm, ln2w, ln2b);     // xt = ln(xt + f)
    rms_rows<<<(int)MROWS_ALL, 128, 0, stream>>>(G_, tnw, G_);                  // rmsnorm in place

    final_add<<<(int)(NJC / 256), 256, 0, stream>>>(G_, F_, x, out_x);
}
